// DIN_81140522156074
// MI455X (gfx1250) — matrix operand order repaired and checked by reading, not compiled, not run on hardware
//
#include <hip/hip_runtime.h>
#include <hip/hip_bf16.h>

// ---------------- problem constants (match reference) ----------------
constexpr int M_    = 200;   // history length
constexpr int E_    = 64;    // embedding
constexpr int ADIM_ = 192;   // 3*E
constexpr int H1_   = 96;    // ADIM/2
constexpr int MT_   = 13;    // ceil(200/16) M-tiles
constexpr int FEA_  = 256;
constexpr int F1_   = 128;   // FEA/2
constexpr int F2_   = 85;    // FEA/3
constexpr int F3_   = 64;    // FEA/4
constexpr int OTHER_= 128;

typedef __attribute__((ext_vector_type(16))) __bf16 v16bf;
typedef __attribute__((ext_vector_type(8)))  float  v8f;
typedef __attribute__((ext_vector_type(4)))  int    v4i_;

union AFrag { v16bf v; unsigned short u[16]; };
union BFrag { v16bf v; uint4 q[2]; };

#define AS1_ __attribute__((address_space(1)))
#define AS3_ __attribute__((address_space(3)))

#if __has_builtin(__builtin_amdgcn_global_load_async_to_lds_b128) && \
    __has_builtin(__builtin_amdgcn_s_wait_asynccnt)
#define USE_ASYNC_LDS 1
#else
#define USE_ASYNC_LDS 0
#endif

__device__ __forceinline__ unsigned short f2bf(float f) {
  union { float f; unsigned int u; } c; c.f = f;
  unsigned int u = c.u;
  u += 0x7FFFu + ((u >> 16) & 1u);   // round-to-nearest-even
  return (unsigned short)(u >> 16);
}

__global__ __launch_bounds__(256, 4)
void din_fused(const float* __restrict__ hist,   const float* __restrict__ target,
               const float* __restrict__ other,
               const float* __restrict__ aW1,    const float* __restrict__ ab1,
               const float* __restrict__ aW2,    const float* __restrict__ ab2,
               const float* __restrict__ oW1,    const float* __restrict__ ob1,
               const float* __restrict__ oW2,    const float* __restrict__ ob2,
               const float* __restrict__ oW3,    const float* __restrict__ ob3,
               const float* __restrict__ fW,     const float* __restrict__ fb,
               float* __restrict__ out)
{
  __shared__ __align__(16) float          s_hist[M_ * E_];        // 51200 B
  __shared__ __align__(16) float          s_tgt[E_];
  __shared__ __align__(16) unsigned short s_w1t[H1_ * ADIM_];     // 36864 B, [n][k] k-contig
  __shared__ __align__(16) float          s_ab1[H1_];
  __shared__ __align__(16) float          s_aw2[H1_];
  __shared__ __align__(16) float          s_aij[MT_ * 16];        // 208
  __shared__ __align__(16) float          s_x[FEA_];
  __shared__ __align__(16) float          s_y[F1_];
  __shared__ __align__(16) float          s_red[256];

  const int tid  = threadIdx.x;
  const int b    = blockIdx.x;
  const int lane = tid & 31;
  const int wave = tid >> 5;

  // ---------------- Phase 0: cooperative LDS fill ----------------
  {
    const float4* src = (const float4*)(hist + (size_t)b * (M_ * E_));
    float4* dst = (float4*)s_hist;
#if USE_ASYNC_LDS
    // CDNA5 async global->LDS: bypasses VGPRs, tracked on ASYNCcnt.
    for (int i = tid; i < (M_ * E_) / 4; i += 256) {
      AS1_ v4i_* g = (AS1_ v4i_*)(unsigned long long)(const void*)(src + i);
      // low 32 bits of a generic LDS pointer == LDS byte offset (ISA 10.2)
      AS3_ v4i_* l = (AS3_ v4i_*)(unsigned long long)
                     (unsigned)(unsigned long long)(const void*)(dst + i);
      __builtin_amdgcn_global_load_async_to_lds_b128(g, l, 0, 0);
    }
    __builtin_amdgcn_s_wait_asynccnt(0);
#else
    for (int i = tid; i < (M_ * E_) / 4; i += 256) dst[i] = src[i];
#endif
  }
  if (tid < E_)  s_tgt[tid] = target[(size_t)b * E_ + tid];
  if (tid < H1_) { s_ab1[tid] = ab1[tid]; s_aw2[tid] = aW2[tid]; }
  // transpose + downconvert aW1 (ADIM x H1, row-major) -> bf16 [n][k]
  for (int idx = tid; idx < H1_ * ADIM_; idx += 256) {
    int n = idx % H1_;
    int k = idx / H1_;
    s_w1t[n * ADIM_ + k] = f2bf(aW1[idx]);   // idx == k*H1_ + n (coalesced read)
  }
  __syncthreads();

  // ---------------- Phase 1: attention GEMM (bf16 WMMA) ----------------
  // A (208x192) = [hist | hist*tgt | tgt], built per-fragment on the fly.
  for (int mt = wave; mt < MT_; mt += 8) {
    const int m0 = mt * 16;
    int row = m0 + (lane & 15);
    const float zf = (row < M_) ? 1.f : 0.f;   // zero pad rows >= 200
    if (row >= M_) row = M_ - 1;
    const float* hrow = &s_hist[row * E_];
    const int hs = (lane >> 4) * 8;            // A layout: half-wave K phase

    AFrag af[6];
#pragma unroll
    for (int ks = 0; ks < 6; ++ks) {
#pragma unroll
      for (int half = 0; half < 2; ++half) {
        const int kk = ks * 32 + hs + half * 16;   // run of 8, never crosses a 64-col region
#pragma unroll
        for (int j = 0; j < 8; ++j) {
          const int kc = kk + j;
          float v;
          if (kc < 64)        v = hrow[kc];
          else if (kc < 128)  v = hrow[kc - 64] * s_tgt[kc - 64];
          else                v = s_tgt[kc - 128];
          af[ks].u[half * 8 + j] = f2bf(v * zf);
        }
      }
    }

    float aijp[8] = {0.f, 0.f, 0.f, 0.f, 0.f, 0.f, 0.f, 0.f};
    const int ncol_l = lane & 15;
    const int koff_b = (lane >> 4) * 8;        // B layout: lanes 16-31 hold K 8..15, 24..31

    // Partial unroll: 2 independent accumulator chains (good WMMA ILP) while
    // keeping VGPR pressure under the launch_bounds(256,4) budget.
#pragma unroll 2
    for (int nt = 0; nt < 6; ++nt) {
      v8f c = {0.f, 0.f, 0.f, 0.f, 0.f, 0.f, 0.f, 0.f};
      const int n = nt * 16 + ncol_l;
      const unsigned short* wrow = &s_w1t[n * ADIM_];
#pragma unroll
      for (int ks = 0; ks < 6; ++ks) {
        BFrag bfr;
        const uint4* wp = (const uint4*)(wrow + ks * 32 + koff_b);
        bfr.q[0] = wp[0];
        bfr.q[1] = wp[2];
        c = __builtin_amdgcn_wmma_f32_16x16x32_bf16(
                false, af[ks].v, false, bfr.v, (short)0, c, false, false);
      }
      // fuse: aij[m] += relu(h + ab1[n]) * aW2[n] over this lane's column n
      const float bias = s_ab1[n];
      const float w2   = s_aw2[n];
#pragma unroll
      for (int r = 0; r < 8; ++r) {
        const float h = fmaxf(c[r] + bias, 0.f);
        aijp[r] += h * w2;
      }
    }

    // reduce over the 16 lanes sharing each M row (C layout: M = r + 8*(lane>>4))
#pragma unroll
    for (int r = 0; r < 8; ++r) {
      float v = aijp[r];
      v += __shfl_xor(v, 1, 32);
      v += __shfl_xor(v, 2, 32);
      v += __shfl_xor(v, 4, 32);
      v += __shfl_xor(v, 8, 32);
      if ((lane & 15) == 0) {
        const int mrow = m0 + r + (lane >> 4) * 8;
        s_aij[mrow] = v;
      }
    }
  }
  __syncthreads();

  // ---------------- Phase 2: tmp[e] = sum_m (aij[m]+ab2) * hist[m][e] ----------------
  {
    const float ab2v  = ab2[0];
    const int   e     = tid & 63;
    const int   chunk = tid >> 6;              // 4 chunks of 50 rows
    const int   mstart = chunk * 50;
    float acc = 0.f;
    for (int m = mstart; m < mstart + 50; ++m)
      acc = fmaf(s_aij[m] + ab2v, s_hist[m * E_ + e], acc);
    s_red[chunk * 64 + e] = acc;
  }
  __syncthreads();
  if (tid < E_) {
    s_x[tid]      = s_red[tid] + s_red[64 + tid] + s_red[128 + tid] + s_red[192 + tid];
    s_x[E_ + tid] = s_tgt[tid];
  }
  if (tid < OTHER_) s_x[128 + tid] = other[(size_t)b * OTHER_ + tid];
  __syncthreads();

  // ---------------- Phase 3: MLP tail ----------------
  if (tid < F1_) {
    float acc = ob1[tid];
    for (int i = 0; i < FEA_; ++i) acc = fmaf(s_x[i], oW1[i * F1_ + tid], acc);
    s_y[tid] = fmaxf(acc, 0.f);
  }
  __syncthreads();
  if (tid < F2_) {
    float acc = ob2[tid];
    for (int i = 0; i < F1_; ++i) acc = fmaf(s_y[i], oW2[i * F2_ + tid], acc);
    s_red[tid] = fmaxf(acc, 0.f);
  }
  __syncthreads();
  if (tid < F3_) {
    float acc = ob3[tid];
    for (int i = 0; i < F2_; ++i) acc = fmaf(s_red[i], oW3[i * F3_ + tid], acc);
    s_y[tid] = fmaxf(acc, 0.f);
  }
  __syncthreads();
  if (tid < F3_) s_x[tid] = s_y[tid] * fW[tid];
  __syncthreads();
  if (tid == 0) {
    float acc = fb[0];
    for (int i = 0; i < F3_; ++i) acc += s_x[i];
    out[b] = 1.f / (1.f + __expf(-acc));
  }
}

extern "C" void kernel_launch(void* const* d_in, const int* in_sizes, int n_in,
                              void* d_out, int out_size, void* d_ws, size_t ws_size,
                              hipStream_t stream) {
  (void)n_in; (void)d_ws; (void)ws_size; (void)out_size;
  const float* hist   = (const float*)d_in[0];
  const float* target = (const float*)d_in[1];
  const float* other  = (const float*)d_in[2];
  const float* aW1    = (const float*)d_in[3];
  const float* ab1    = (const float*)d_in[4];
  const float* aW2    = (const float*)d_in[5];
  const float* ab2    = (const float*)d_in[6];
  const float* oW1    = (const float*)d_in[7];
  const float* ob1    = (const float*)d_in[8];
  const float* oW2    = (const float*)d_in[9];
  const float* ob2    = (const float*)d_in[10];
  const float* oW3    = (const float*)d_in[11];
  const float* ob3    = (const float*)d_in[12];
  const float* fW     = (const float*)d_in[13];
  const float* fb     = (const float*)d_in[14];
  float* out = (float*)d_out;

  const int B = in_sizes[1] / E_;   // 8192
  din_fused<<<dim3(B), dim3(256), 0, stream>>>(
      hist, target, other, aW1, ab1, aW2, ab2,
      oW1, ob1, oW2, ob2, oW3, ob3, fW, fb, out);
}
